// RNN_GAUSS_66486093742746
// MI455X (gfx1250) — compile-verified, matrix operand order checked by reading, not run on hardware
//
#include <hip/hip_runtime.h>
#include <hip/hip_bf16.h>
#include <math.h>

// Problem constants (match reference)
#define A_N   10
#define B_N   1024
#define Y_N   94
#define R_N   256
#define H_N   128
#define T_N   50
#define G3    768   // 3*R
#define KP    352   // padded enc K: 96 (y padded) + 256 (h)
#define KY    96
#define RPW   32    // batch rows per wave = 2 M-tiles (register blocking)

typedef __attribute__((ext_vector_type(16))) __bf16 v16bf;
typedef __attribute__((ext_vector_type(8)))  float  v8f;

union FragU { v16bf v; uint4 q[2]; };

// Load a 16x32 bf16 A-frag (or 32x16 B-frag) in the CDNA5 WMMA lane layout:
// lane L holds row (L&15); K = kOff + 8*(L>>4) + [0..8) and +16.
__device__ __forceinline__ v16bf load_frag(const __bf16* p, int stride, int lane, int kOff) {
  const int row = lane & 15;
  const int k0  = kOff + ((lane >> 4) << 3);
  FragU f;
  f.q[0] = *(const uint4*)(p + (size_t)row * stride + k0);
  f.q[1] = *(const uint4*)(p + (size_t)row * stride + k0 + 16);
  return f.v;
}

__device__ __forceinline__ v8f wmma_bf16(v16bf a, v16bf b, v8f c) {
  return __builtin_amdgcn_wmma_f32_16x16x32_bf16(false, a, false, b, (short)0, c,
                                                 false, false);
}

__device__ __forceinline__ v8f splat8(float x) { v8f r = {x,x,x,x,x,x,x,x}; return r; }
__device__ __forceinline__ float sigmoidf_(float x) { return 1.f / (1.f + __expf(-x)); }
// tanh via the transcendental unit: tanh(x) = 2*sigmoid(2x) - 1
__device__ __forceinline__ float tanhf_(float x) {
  float s = 1.f / (1.f + __expf(-2.f * x));
  return 2.f * s - 1.f;
}
__device__ __forceinline__ float softplusf_(float x) {
  return (x > 20.f) ? x : log1pf(__expf(x));
}

#define DS_FENCE() asm volatile("s_wait_dscnt 0" ::: "memory")

// ---------------- prep kernels ----------------

__global__ void zero_acc_k(float* acc) {
  if (threadIdx.x < 4) acc[threadIdx.x] = 0.f;
}

__global__ void pack_wih_k(const float* __restrict__ w, __bf16* __restrict__ out) {
  int idx = blockIdx.x * 256 + threadIdx.x;
  const int total = A_N * G3 * KP;
  if (idx >= total) return;
  int k = idx % KP;
  int g = (idx / KP) % G3;
  int a = idx / (KP * G3);
  float v = 0.f;
  if (k < 94)       v = w[((size_t)a * G3 + g) * 350 + k];
  else if (k >= KY) v = w[((size_t)a * G3 + g) * 350 + 94 + (k - KY)];
  out[idx] = (__bf16)v;
}

__global__ void cvt_bf16_k(const float* __restrict__ in, __bf16* __restrict__ out, int n) {
  int idx = blockIdx.x * 256 + threadIdx.x;
  if (idx < n) out[idx] = (__bf16)in[idx];
}

__global__ void pack_msw_k(const float* __restrict__ mw, const float* __restrict__ sw,
                           __bf16* __restrict__ out) {
  int idx = blockIdx.x * 256 + threadIdx.x;
  const int total = A_N * 16 * H_N;
  if (idx >= total) return;
  int k = idx % H_N;
  int r = (idx / H_N) % 16;
  int a = idx / (H_N * 16);
  float v = 0.f;
  if (r < 2)      v = mw[((size_t)a * 2 + r) * H_N + k];
  else if (r < 4) v = sw[((size_t)a * 2 + (r - 2)) * H_N + k];
  out[idx] = (__bf16)v;
}

__global__ void finalize_k(const float* acc, float* out) {
  if (threadIdx.x < 3) out[threadIdx.x] = acc[threadIdx.x] / (float)(T_N * A_N);
}

// ---------------- main kernel ----------------
// 1 wave (32 threads) per block; each wave owns 32 batch rows (two 16-row
// M-tiles) of one agent and runs all T=50 steps locally. Every B-fragment
// (weights, from L2) feeds TWO WMMAs; A-fragments are re-read from LDS
// for every output tile. grid = A_N * (B_N / RPW) = 320 blocks.

__global__ __launch_bounds__(32)
void rnn_main_k(const float* __restrict__ states,
                const float* __restrict__ b_ih, const float* __restrict__ b_hh,
                const float* __restrict__ d1_b, const float* __restrict__ d2_b,
                const float* __restrict__ m_b,  const float* __restrict__ s_b,
                const __bf16* __restrict__ wihP, const __bf16* __restrict__ whhP,
                const __bf16* __restrict__ d1wP, const __bf16* __restrict__ d2wP,
                const __bf16* __restrict__ mswP,
                float* __restrict__ acc) {
  __shared__ __bf16 sH[2][RPW * R_N];   // h double buffer (bf16)       32 KB
  __shared__ __bf16 sY[RPW * KY];       // staged y_t (bf16, K=96 pad)   6 KB
  __shared__ __bf16 sD1[RPW * H_N];     // relu(d1) activations          8 KB
  __shared__ __bf16 sD2[RPW * H_N];     // relu(d2) activations          8 KB
  __shared__ float  sStats[RPW * 4];    // raw mean/std-preact columns  .5 KB

  const int lane = threadIdx.x & 31;
  const int a    = blockIdx.x >> 5;            // 32 blocks per agent
  const int b0   = (blockIdx.x & 31) * RPW;

  const __bf16* Wih  = wihP + (size_t)a * G3 * KP;
  const __bf16* Whh  = whhP + (size_t)a * G3 * R_N;
  const __bf16* D1w  = d1wP + (size_t)a * H_N * R_N;
  const __bf16* D2w  = d2wP + (size_t)a * H_N * H_N;
  const __bf16* Msw  = mswP + (size_t)a * 16 * H_N;

  // h0 = 0
  for (int i = lane; i < RPW * R_N; i += 32) sH[0][i] = (__bf16)0.f;
  DS_FENCE();

  const float LOG2PI = 1.8378770664093453f;
  float accL = 0.f, accEp = 0.f, accEv = 0.f;
  int cur = 0;

  for (int t = 0; t < T_N; ++t) {
    // ---- stage y_t into LDS as bf16 (one row per lane, vectorized) ----
    {
      const int r = lane;
      const float2* src =
          (const float2*)(states + (((size_t)t * A_N + a) * B_N + (b0 + r)) * Y_N);
      uint* dst = (uint*)&sY[r * KY];
      for (int j = 0; j < 47; ++j) {           // 94 floats = 47 float2
        float2 v = src[j];
        union { __bf16 h[2]; uint u; } p;
        p.h[0] = (__bf16)v.x; p.h[1] = (__bf16)v.y;
        dst[j] = p.u;
      }
      dst[47] = 0u;                            // zero-pad cols 94,95
    }
    DS_FENCE();

    const __bf16* hc = sH[cur];
    __bf16*       hn = sH[cur ^ 1];
    const int col16 = lane & 15;
    const int m0    = (lane >> 4) << 3;

    // ---- d1 = relu(h @ d1_w^T + d1_b): 32x128 ----
    for (int nt = 0; nt < 8; ++nt) {
      const float bias = d1_b[a * H_N + nt * 16 + col16];
      v8f dacc0 = splat8(bias), dacc1 = splat8(bias);
      const __bf16* bp = D1w + (size_t)(nt * 16) * R_N;
#pragma unroll
      for (int kt = 0; kt < 8; ++kt) {
        v16bf bf = load_frag(bp, R_N, lane, kt * 32);
        dacc0 = wmma_bf16(load_frag(hc,             R_N, lane, kt * 32), bf, dacc0);
        dacc1 = wmma_bf16(load_frag(hc + 16 * R_N,  R_N, lane, kt * 32), bf, dacc1);
      }
      const int c = nt * 16 + col16;
#pragma unroll
      for (int v = 0; v < 8; ++v) {
        float x0 = dacc0[v]; x0 = x0 > 0.f ? x0 : 0.f;
        float x1 = dacc1[v]; x1 = x1 > 0.f ? x1 : 0.f;
        sD1[(m0 + v) * H_N + c]      = (__bf16)x0;
        sD1[(16 + m0 + v) * H_N + c] = (__bf16)x1;
      }
    }
    DS_FENCE();

    // ---- d2 = relu(d1 @ d2_w^T + d2_b): 32x128 ----
    for (int nt = 0; nt < 8; ++nt) {
      const float bias = d2_b[a * H_N + nt * 16 + col16];
      v8f dacc0 = splat8(bias), dacc1 = splat8(bias);
      const __bf16* bp = D2w + (size_t)(nt * 16) * H_N;
#pragma unroll
      for (int kt = 0; kt < 4; ++kt) {
        v16bf bf = load_frag(bp, H_N, lane, kt * 32);
        dacc0 = wmma_bf16(load_frag(sD1,             H_N, lane, kt * 32), bf, dacc0);
        dacc1 = wmma_bf16(load_frag(sD1 + 16 * H_N,  H_N, lane, kt * 32), bf, dacc1);
      }
      const int c = nt * 16 + col16;
#pragma unroll
      for (int v = 0; v < 8; ++v) {
        float x0 = dacc0[v]; x0 = x0 > 0.f ? x0 : 0.f;
        float x1 = dacc1[v]; x1 = x1 > 0.f ? x1 : 0.f;
        sD2[(m0 + v) * H_N + c]      = (__bf16)x0;
        sD2[(16 + m0 + v) * H_N + c] = (__bf16)x1;
      }
    }
    DS_FENCE();

    // ---- mean/std pre-activations: two 16x16 tiles (cols 0,1 mean; 2,3 std) ----
    {
      v8f macc0 = splat8(0.f), macc1 = splat8(0.f);
#pragma unroll
      for (int kt = 0; kt < 4; ++kt) {
        v16bf bf = load_frag(Msw, H_N, lane, kt * 32);
        macc0 = wmma_bf16(load_frag(sD2,             H_N, lane, kt * 32), bf, macc0);
        macc1 = wmma_bf16(load_frag(sD2 + 16 * H_N,  H_N, lane, kt * 32), bf, macc1);
      }
      if (col16 < 4) {
#pragma unroll
        for (int v = 0; v < 8; ++v) {
          sStats[(m0 + v) * 4 + col16]      = macc0[v];
          sStats[(16 + m0 + v) * 4 + col16] = macc1[v];
        }
      }
    }
    DS_FENCE();

    // ---- per-row scalar stats (one batch row per lane) ----
    {
      const int r = lane;
      const int brow = b0 + r;
      float mx = sStats[r * 4 + 0] + m_b[a * 2 + 0];
      float my = sStats[r * 4 + 1] + m_b[a * 2 + 1];
      float sx = softplusf_(sStats[r * 4 + 2] + s_b[a * 2 + 0]);
      float sy = softplusf_(sStats[r * 4 + 3] + s_b[a * 2 + 1]);
      const float* f_t  = states + (((size_t)t       * A_N + a) * B_N + brow) * Y_N + 4 * a;
      const float* f_t1 = states + (((size_t)(t + 1) * A_N + a) * B_N + brow) * Y_N + 4 * a;
      float x0 = f_t1[2], x1 = f_t1[3];
      float e0 = (x0 - mx) / sx, e1 = (x1 - my) / sy;
      accL += 0.5f * (e0 * e0 + e1 * e1 + 2.f * (__logf(sx) + __logf(sy)) + 2.f * LOG2PI);
      float np0 = f_t[0] + f_t[2] * 0.1f, np1 = f_t[1] + f_t[3] * 0.1f;
      float g0 = f_t1[0], g1 = f_t1[1];
      accEp += sqrtf((np0 - g0) * (np0 - g0) + (np1 - g1) * (np1 - g1));
      accEv += sqrtf((mx - x0) * (mx - x0) + (my - x1) * (my - x1));
    }

    // ---- GRU gates, fused gi+gh over 16 column tiles of R ----
    // 8 independent accumulator chains; each B-frag feeds 2 WMMAs.
    for (int ct = 0; ct < 16; ++ct) {
      const int col = ct * 16 + col16;
      const float br  = b_ih[a * G3 + col]           + b_hh[a * G3 + col];
      const float bz  = b_ih[a * G3 + R_N + col]     + b_hh[a * G3 + R_N + col];
      const float bin = b_ih[a * G3 + 2 * R_N + col];
      const float bhn = b_hh[a * G3 + 2 * R_N + col];
      v8f ar0 = splat8(br),  ar1 = splat8(br);
      v8f az0 = splat8(bz),  az1 = splat8(bz);
      v8f an0 = splat8(bin), an1 = splat8(bin);
      v8f ah0 = splat8(bhn), ah1 = splat8(bhn);
      const __bf16* WiR = Wih + (size_t)(ct * 16)           * KP;
      const __bf16* WiZ = Wih + (size_t)(R_N + ct * 16)     * KP;
      const __bf16* WiN = Wih + (size_t)(2 * R_N + ct * 16) * KP;
      const __bf16* WhR = Whh + (size_t)(ct * 16)           * R_N;
      const __bf16* WhZ = Whh + (size_t)(R_N + ct * 16)     * R_N;
      const __bf16* WhN = Whh + (size_t)(2 * R_N + ct * 16) * R_N;
      // y-part of enc (K = 0..95, padded)
#pragma unroll
      for (int kt = 0; kt < 3; ++kt) {
        v16bf a0 = load_frag(sY,            KY, lane, kt * 32);
        v16bf a1 = load_frag(sY + 16 * KY,  KY, lane, kt * 32);
        v16bf bR = load_frag(WiR, KP, lane, kt * 32);
        ar0 = wmma_bf16(a0, bR, ar0);  ar1 = wmma_bf16(a1, bR, ar1);
        v16bf bZ = load_frag(WiZ, KP, lane, kt * 32);
        az0 = wmma_bf16(a0, bZ, az0);  az1 = wmma_bf16(a1, bZ, az1);
        v16bf bN = load_frag(WiN, KP, lane, kt * 32);
        an0 = wmma_bf16(a0, bN, an0);  an1 = wmma_bf16(a1, bN, an1);
      }
      // h-part of enc (K = 96..351) + recurrent weights
#pragma unroll
      for (int kt = 0; kt < 8; ++kt) {
        v16bf a0 = load_frag(hc,            R_N, lane, kt * 32);
        v16bf a1 = load_frag(hc + 16 * R_N, R_N, lane, kt * 32);
        v16bf b0f = load_frag(WiR, KP, lane, KY + kt * 32);
        ar0 = wmma_bf16(a0, b0f, ar0);  ar1 = wmma_bf16(a1, b0f, ar1);
        v16bf b1f = load_frag(WhR, R_N, lane, kt * 32);
        ar0 = wmma_bf16(a0, b1f, ar0);  ar1 = wmma_bf16(a1, b1f, ar1);
        v16bf b2f = load_frag(WiZ, KP, lane, KY + kt * 32);
        az0 = wmma_bf16(a0, b2f, az0);  az1 = wmma_bf16(a1, b2f, az1);
        v16bf b3f = load_frag(WhZ, R_N, lane, kt * 32);
        az0 = wmma_bf16(a0, b3f, az0);  az1 = wmma_bf16(a1, b3f, az1);
        v16bf b4f = load_frag(WiN, KP, lane, KY + kt * 32);
        an0 = wmma_bf16(a0, b4f, an0);  an1 = wmma_bf16(a1, b4f, an1);
        v16bf b5f = load_frag(WhN, R_N, lane, kt * 32);
        ah0 = wmma_bf16(a0, b5f, ah0);  ah1 = wmma_bf16(a1, b5f, ah1);
      }
      // elementwise GRU update in D-layout, write to the other h buffer
#pragma unroll
      for (int v = 0; v < 8; ++v) {
        {
          float r = sigmoidf_(ar0[v]);
          float z = sigmoidf_(az0[v]);
          float n = tanhf_(an0[v] + r * ah0[v]);
          float ho = (float)hc[(m0 + v) * R_N + col];
          hn[(m0 + v) * R_N + col] = (__bf16)((1.f - z) * n + z * ho);
        }
        {
          float r = sigmoidf_(ar1[v]);
          float z = sigmoidf_(az1[v]);
          float n = tanhf_(an1[v] + r * ah1[v]);
          float ho = (float)hc[(16 + m0 + v) * R_N + col];
          hn[(16 + m0 + v) * R_N + col] = (__bf16)((1.f - z) * n + z * ho);
        }
      }
    }
    DS_FENCE();
    cur ^= 1;
  }

  // ---- wave reduction + global atomic flush ----
  float vL = accL, vEp = accEp, vEv = accEv;
  for (int off = 16; off > 0; off >>= 1) {
    vL  += __shfl_down(vL, off, 32);
    vEp += __shfl_down(vEp, off, 32);
    vEv += __shfl_down(vEv, off, 32);
  }
  if (lane == 0) {
    atomicAdd(&acc[0], vL);
    atomicAdd(&acc[1], vEp);
    atomicAdd(&acc[2], vEv);
  }
}

// ---------------- launch ----------------

extern "C" void kernel_launch(void* const* d_in, const int* in_sizes, int n_in,
                              void* d_out, int out_size, void* d_ws, size_t ws_size,
                              hipStream_t stream) {
  const float* states = (const float*)d_in[0];
  const float* W_ih   = (const float*)d_in[1];
  const float* W_hh   = (const float*)d_in[2];
  const float* b_ih   = (const float*)d_in[3];
  const float* b_hh   = (const float*)d_in[4];
  const float* d1_w   = (const float*)d_in[5];
  const float* d1_b   = (const float*)d_in[6];
  const float* d2_w   = (const float*)d_in[7];
  const float* d2_b   = (const float*)d_in[8];
  const float* m_w    = (const float*)d_in[9];
  const float* m_b    = (const float*)d_in[10];
  const float* s_w    = (const float*)d_in[11];
  const float* s_b    = (const float*)d_in[12];
  (void)in_sizes; (void)n_in; (void)out_size; (void)ws_size;

  char*   ws    = (char*)d_ws;
  float*  acc   = (float*)ws;
  __bf16* wih_p = (__bf16*)(ws + 256);
  __bf16* whh_p = wih_p + (size_t)A_N * G3 * KP;
  __bf16* d1w_p = whh_p + (size_t)A_N * G3 * R_N;
  __bf16* d2w_p = d1w_p + (size_t)A_N * H_N * R_N;
  __bf16* msw_p = d2w_p + (size_t)A_N * H_N * H_N;

  zero_acc_k<<<1, 32, 0, stream>>>(acc);
  { int n = A_N * G3 * KP;  pack_wih_k<<<(n + 255) / 256, 256, 0, stream>>>(W_ih, wih_p); }
  { int n = A_N * G3 * R_N; cvt_bf16_k<<<(n + 255) / 256, 256, 0, stream>>>(W_hh, whh_p, n); }
  { int n = A_N * H_N * R_N; cvt_bf16_k<<<(n + 255) / 256, 256, 0, stream>>>(d1_w, d1w_p, n); }
  { int n = A_N * H_N * H_N; cvt_bf16_k<<<(n + 255) / 256, 256, 0, stream>>>(d2_w, d2w_p, n); }
  { int n = A_N * 16 * H_N; pack_msw_k<<<(n + 255) / 256, 256, 0, stream>>>(m_w, s_w, msw_p); }

  rnn_main_k<<<A_N * (B_N / RPW), 32, 0, stream>>>(
      states, b_ih, b_hh, d1_b, d2_b, m_b, s_b,
      wih_p, whh_p, d1w_p, d2w_p, msw_p, acc);

  finalize_k<<<1, 32, 0, stream>>>(acc, (float*)d_out);
}
